// WindowContextAttention_2001454760425
// MI455X (gfx1250) — matrix operand order repaired and checked by reading, not compiled, not run on hardware
//
#include <hip/hip_runtime.h>

// ---------------------------------------------------------------------------
// SwinV2 window context attention for MI455X (gfx1250, wave32, WMMA).
// One workgroup (128 threads = 4 waves) per window; wave w owns head w.
// All GEMMs via v_wmma_f32_16x16x32_f16, f32 accumulation.
// ---------------------------------------------------------------------------

typedef __attribute__((ext_vector_type(16))) _Float16 v16h;
typedef __attribute__((ext_vector_type(8)))  _Float16 v8h;
typedef __attribute__((ext_vector_type(8)))  float    v8f;

#define WMMA_F16(a, b, c) \
  __builtin_amdgcn_wmma_f32_16x16x32_f16(false, (a), false, (b), (short)0, (c), false, false)

#define NTOK 64
#define DIM  128
#define NH   4
#define HD   32
#define NWIN 64
#define BWIN 4096

// ---- workspace layout (bytes) ----
#define WS_GWQ   0u          // f16 [128][128]
#define WS_GWK   32768u      // f16 [128][128]   (kv_w rows 0..127)
#define WS_GWV   65536u      // f16 [128][128]   (kv_w rows 128..255)
#define WS_GWP   98304u      // f16 [128][128]
#define WS_SCALE 131072u     // f32 [4]
#define WS_TBL   131136u     // f32 [225][4]
#define WS_BIAS  134784u     // f32 [4][64][64]

// ---- dynamic LDS layout (bytes) ----
// [0      ,16384) sQ   f16 [64][128]   (later: P probs, wave w at +w*8KB over sQ+sK)
// [16384  ,32768) sK   f16 [64][128]
// [32768  ,49152) sVt  f16 [128][64]   (V transposed: [channel][token])
// [49152  ,65536) sX   f16 [64][128]   \
// [65536  ,81920) sC   f16 [64][128]    > phase-1 use of R1
// [81920 ,114688) sF   f32 [64][128]   /
// [49152 ,114688) sS   f32 [4][64][64] (scores, phase 4-5)
// [49152 , 81920) sO   f32 [64][128]   (attention output, phase 6-7)
#define SMEM_BYTES 114688

// ---------------------------------------------------------------------------
// Fragment loaders (CDNA5 WMMA VGPR layouts, ISA 7.12.2)
// ---------------------------------------------------------------------------

// A: 16x32 f16 tile, row-major source, ld in f16 elements.
// lanes 0-15 : row = lane, K = {0..7, 16..23}
// lanes 16-31: row = lane-16, K = {8..15, 24..31}
__device__ __forceinline__ v16h load_a16(const _Float16* __restrict__ base, int ld, int lane) {
  const int r  = lane & 15;
  const int ks = (lane >> 4) << 3;                // 0 or 8
  const _Float16* p = base + r * ld + ks;
  v8h lo = *(const v8h*)p;
  v8h hi = *(const v8h*)(p + 16);
  v16h a;
#pragma unroll
  for (int i = 0; i < 8; ++i) { a[i] = lo[i]; a[i + 8] = hi[i]; }
  return a;
}

// A tile from an f32 source (convert on the fly).
__device__ __forceinline__ v16h load_a32(const float* __restrict__ base, int ld, int lane) {
  const int r  = lane & 15;
  const int ks = (lane >> 4) << 3;
  const float* p = base + r * ld + ks;
  v16h a;
#pragma unroll
  for (int i = 0; i < 8; ++i) {
    a[i]     = (_Float16)p[i];
    a[i + 8] = (_Float16)p[16 + i];
  }
  return a;
}

// B: 32x16 f16 tile; source stored [N][K] row-major (weight layout W[out][in]).
// lanes 0-15 : col = lane,  K = {0..7, 16..23} ; lanes 16-31: col = lane-16, K = {8..15, 24..31}
__device__ __forceinline__ v16h load_b16(const _Float16* __restrict__ base, int ld, int lane) {
  const int n  = lane & 15;
  const int ks = (lane >> 4) << 3;                // 0 or 8
  const _Float16* p = base + n * ld + ks;
  v8h lo = *(const v8h*)p;
  v8h hi = *(const v8h*)(p + 16);
  v16h b;
#pragma unroll
  for (int i = 0; i < 8; ++i) { b[i] = lo[i]; b[i + 8] = hi[i]; }
  return b;
}

// D: 16x16 f32; element (m,n): lane = n + 16*(m>=8), vgpr = m%8
__device__ __forceinline__ void store_d(float* __restrict__ base, int ld, int lane, v8f d) {
  const int n  = lane & 15;
  const int m0 = (lane >> 4) << 3;
#pragma unroll
  for (int r = 0; r < 8; ++r) base[(m0 + r) * ld + n] = d[r];
}

// 64x32 GEMM slice: D[:, 32w:32w+32] = A(64x128,f16) * W(sub 32 rows of [128][128])^T
__device__ __forceinline__ void gemm_qkv(const _Float16* __restrict__ A,
                                         const _Float16* __restrict__ W,
                                         int w, int lane, v8f (&acc)[4][2]) {
  v8f z = {};
#pragma unroll
  for (int mt = 0; mt < 4; ++mt) { acc[mt][0] = z; acc[mt][1] = z; }
#pragma unroll
  for (int kk = 0; kk < 4; ++kk) {
    const int k0 = kk * 32;
    v16h b0 = load_b16(W + (32 * w +  0) * DIM + k0, DIM, lane);
    v16h b1 = load_b16(W + (32 * w + 16) * DIM + k0, DIM, lane);
#pragma unroll
    for (int mt = 0; mt < 4; ++mt) {
      v16h a = load_a16(A + mt * 16 * DIM + k0, DIM, lane);
      acc[mt][0] = WMMA_F16(a, b0, acc[mt][0]);
      acc[mt][1] = WMMA_F16(a, b1, acc[mt][1]);
    }
  }
}

// per-(row,head) L2 normalization: src f32 [64][128] -> dst f16 [64][128]
__device__ __forceinline__ void l2norm_to(const float* __restrict__ src,
                                          _Float16* __restrict__ dst, int tid) {
  for (int p = tid; p < NTOK * NH; p += 128) {
    const int row = p >> 2, h = p & 3;
    const float* s = src + row * DIM + h * HD;
    float ss = 0.f;
#pragma unroll
    for (int j = 0; j < HD; ++j) ss += s[j] * s[j];
    const float inv = 1.f / fmaxf(sqrtf(ss), 1e-12f);
    _Float16* d = dst + row * DIM + h * HD;
#pragma unroll
    for (int j = 0; j < HD; ++j) d[j] = (_Float16)(s[j] * inv);
  }
}

// ---------------------------------------------------------------------------
// Prep kernels (run once per launch; tiny)
// ---------------------------------------------------------------------------

__global__ void prep_weights_kernel(const float* __restrict__ qw,
                                    const float* __restrict__ kvw,
                                    const float* __restrict__ pw,
                                    _Float16* __restrict__ gq, _Float16* __restrict__ gk,
                                    _Float16* __restrict__ gv, _Float16* __restrict__ gp) {
  int i = blockIdx.x * blockDim.x + threadIdx.x;
  const int stride = gridDim.x * blockDim.x;
  for (; i < 65536; i += stride) {
    if (i < 16384) {
      gq[i] = (_Float16)qw[i];
    } else if (i < 49152) {
      const int j = i - 16384;
      const float v = kvw[j];
      if (j < 16384) gk[j] = (_Float16)v;
      else           gv[j - 16384] = (_Float16)v;
    } else {
      gp[i - 49152] = (_Float16)pw[i - 49152];
    }
  }
}

__global__ void prep_cpb_kernel(const float* __restrict__ tab,   // (1,15,15,2)
                                const float* __restrict__ w1,    // (512,2)
                                const float* __restrict__ b1,    // (512,)
                                const float* __restrict__ w2,    // (4,512)
                                const float* __restrict__ ls,    // (4,1,1)
                                float* __restrict__ tblw,        // [225][4]
                                float* __restrict__ scaleH) {    // [4]
  const int t = threadIdx.x;
  if (t < 225) {
    const float cx = tab[2 * t], cy = tab[2 * t + 1];
    float a0 = 0.f, a1 = 0.f, a2 = 0.f, a3 = 0.f;
    for (int j = 0; j < 512; ++j) {
      const float h = fmaxf(w1[2 * j] * cx + w1[2 * j + 1] * cy + b1[j], 0.f);
      a0 += w2[j]        * h;
      a1 += w2[512 + j]  * h;
      a2 += w2[1024 + j] * h;
      a3 += w2[1536 + j] * h;
    }
    tblw[4 * t + 0] = a0; tblw[4 * t + 1] = a1;
    tblw[4 * t + 2] = a2; tblw[4 * t + 3] = a3;
  }
  if (t < 4) scaleH[t] = __expf(fminf(ls[t], 4.6051702f));   // log(1/0.01)
}

__global__ void prep_bias_kernel(const float* __restrict__ tblw,
                                 const int* __restrict__ idx,    // (64,64)
                                 float* __restrict__ biasX) {    // [4][64][64]
  const int i = blockIdx.x * blockDim.x + threadIdx.x;
  if (i >= NH * NTOK * NTOK) return;
  const int h = i >> 12;
  const int rem = i & 4095;
  const float v = tblw[4 * idx[rem] + h];
  biasX[i] = 16.f / (1.f + __expf(-v));
}

// ---------------------------------------------------------------------------
// Main attention kernel: one block per window
// ---------------------------------------------------------------------------

__global__ __launch_bounds__(128) void swin_attn_kernel(
    const float* __restrict__ x, const float* __restrict__ ctx,
    const float* __restrict__ mask,
    const float* __restrict__ q_b, const float* __restrict__ v_b,
    const _Float16* __restrict__ gwq, const _Float16* __restrict__ gwk,
    const _Float16* __restrict__ gwv, const _Float16* __restrict__ gwp,
    const float* __restrict__ proj_b,
    const float* __restrict__ scaleH, const float* __restrict__ biasX,
    float* __restrict__ out) {
  extern __shared__ char smem[];
  _Float16* sQ  = (_Float16*)(smem + 0);
  _Float16* sK  = (_Float16*)(smem + 16384);
  _Float16* sVt = (_Float16*)(smem + 32768);
  _Float16* sX  = (_Float16*)(smem + 49152);
  _Float16* sC  = (_Float16*)(smem + 65536);
  float*    sF  = (float*)(smem + 81920);
  float*    sS  = (float*)(smem + 49152);
  float*    sO  = (float*)(smem + 49152);
  _Float16* sP  = (_Float16*)(smem + 0);

  const int tid  = threadIdx.x;
  const int lane = tid & 31;
  const int w    = tid >> 5;          // wave == head
  const int win  = blockIdx.x;
  const int wi   = win & (NWIN - 1);  // window-in-image for shift mask
  const size_t base = (size_t)win * (NTOK * DIM);

  // ---- phase 0: stage x, context as f16 in LDS ----
  {
    const float4* xp = (const float4*)(x + base);
    const float4* cp = (const float4*)(ctx + base);
    for (int i = tid; i < (NTOK * DIM) / 4; i += 128) {
      const float4 a = xp[i];
      const float4 b = cp[i];
      sX[4 * i + 0] = (_Float16)a.x; sX[4 * i + 1] = (_Float16)a.y;
      sX[4 * i + 2] = (_Float16)a.z; sX[4 * i + 3] = (_Float16)a.w;
      sC[4 * i + 0] = (_Float16)b.x; sC[4 * i + 1] = (_Float16)b.y;
      sC[4 * i + 2] = (_Float16)b.z; sC[4 * i + 3] = (_Float16)b.w;
    }
  }
  __syncthreads();

  // ---- phase 1: Q = X @ Wq^T + q_b -> sF (f32) ----
  {
    v8f acc[4][2];
    gemm_qkv(sX, gwq, w, lane, acc);
#pragma unroll
    for (int mt = 0; mt < 4; ++mt)
#pragma unroll
      for (int nt = 0; nt < 2; ++nt) {
        const float bb = q_b[32 * w + nt * 16 + (lane & 15)];
#pragma unroll
        for (int r = 0; r < 8; ++r) acc[mt][nt][r] += bb;
        store_d(sF + mt * 16 * DIM + 32 * w + nt * 16, DIM, lane, acc[mt][nt]);
      }
  }
  __syncthreads();
  l2norm_to(sF, sQ, tid);   // per-head L2 norm -> sQ f16
  __syncthreads();

  // ---- phase 2: K = C @ Wk^T (no bias) -> sF, normalize -> sK ----
  {
    v8f acc[4][2];
    gemm_qkv(sC, gwk, w, lane, acc);
#pragma unroll
    for (int mt = 0; mt < 4; ++mt)
#pragma unroll
      for (int nt = 0; nt < 2; ++nt)
        store_d(sF + mt * 16 * DIM + 32 * w + nt * 16, DIM, lane, acc[mt][nt]);
  }
  __syncthreads();
  l2norm_to(sF, sK, tid);
  __syncthreads();

  // ---- phase 3: V = C @ Wv^T + v_b -> sVt (f16, transposed [chan][tok]) ----
  {
    v8f acc[4][2];
    gemm_qkv(sC, gwv, w, lane, acc);
#pragma unroll
    for (int mt = 0; mt < 4; ++mt)
#pragma unroll
      for (int nt = 0; nt < 2; ++nt) {
        const int n  = 32 * w + nt * 16 + (lane & 15);
        const float bb = v_b[n];
        const int m0 = mt * 16 + ((lane >> 4) << 3);
#pragma unroll
        for (int r = 0; r < 8; ++r)
          sVt[n * NTOK + m0 + r] = (_Float16)(acc[mt][nt][r] + bb);
      }
  }
  __syncthreads();

  // ---- phase 4: S_h = Qn_h @ Kn_h^T (64x64, K=32 single WMMA per tile) ----
  {
    float* sSw = sS + w * (NTOK * NTOK);
#pragma unroll
    for (int nt = 0; nt < 4; ++nt) {
      v16h bk = load_b16(sK + nt * 16 * DIM + 32 * w, DIM, lane);
#pragma unroll
      for (int mt = 0; mt < 4; ++mt) {
        v16h a = load_a16(sQ + mt * 16 * DIM + 32 * w, DIM, lane);
        v8f z = {};
        v8f d = WMMA_F16(a, bk, z);
        store_d(sSw + mt * 16 * NTOK + nt * 16, NTOK, lane, d);
      }
    }
  }
  __syncthreads();

  // ---- phase 5: scale + bias + mask + softmax; probs -> sP (f16) ----
  {
    const float scale = scaleH[w];
    float*    sSw = sS + w * (NTOK * NTOK);
    _Float16* sPw = sP + w * (NTOK * NTOK);
#pragma unroll
    for (int rr = 0; rr < 2; ++rr) {
      const int r = lane + 32 * rr;
      float* row = sSw + r * NTOK;
      const float* br = biasX + (w * NTOK + r) * NTOK;
      const float* mr = mask + ((size_t)wi * NTOK + r) * NTOK;
      float mx = -1e30f;
      for (int c = 0; c < NTOK; ++c) {
        const float v = row[c] * scale + br[c] + mr[c];
        row[c] = v;
        mx = fmaxf(mx, v);
      }
      float s = 0.f;
      for (int c = 0; c < NTOK; ++c) {
        const float e = __expf(row[c] - mx);
        row[c] = e;
        s += e;
      }
      const float inv = 1.f / s;
      _Float16* pr = sPw + r * NTOK;
      for (int c = 0; c < NTOK; ++c) pr[c] = (_Float16)(row[c] * inv);
    }
  }
  __syncthreads();

  // ---- phase 6: O_h = P_h @ V_h (64x32, K=64) ----
  v8f oacc[4][2];
  {
    v8f z = {};
#pragma unroll
    for (int mt = 0; mt < 4; ++mt) { oacc[mt][0] = z; oacc[mt][1] = z; }
    const _Float16* sPw = sP + w * (NTOK * NTOK);
#pragma unroll
    for (int kk = 0; kk < 2; ++kk) {
      const int k0 = kk * 32;
      v16h b0 = load_b16(sVt + (32 * w +  0) * NTOK + k0, NTOK, lane);
      v16h b1 = load_b16(sVt + (32 * w + 16) * NTOK + k0, NTOK, lane);
#pragma unroll
      for (int mt = 0; mt < 4; ++mt) {
        v16h a = load_a16(sPw + mt * 16 * NTOK + k0, NTOK, lane);
        oacc[mt][0] = WMMA_F16(a, b0, oacc[mt][0]);
        oacc[mt][1] = WMMA_F16(a, b1, oacc[mt][1]);
      }
    }
#pragma unroll
    for (int mt = 0; mt < 4; ++mt)
#pragma unroll
      for (int nt = 0; nt < 2; ++nt)
        store_d(sO + mt * 16 * DIM + 32 * w + nt * 16, DIM, lane, oacc[mt][nt]);
  }
  __syncthreads();

  // ---- phase 7: out = O @ Wp^T + proj_b -> global ----
  {
    v8f acc[4][2];
    v8f z = {};
#pragma unroll
    for (int mt = 0; mt < 4; ++mt) { acc[mt][0] = z; acc[mt][1] = z; }
#pragma unroll
    for (int kk = 0; kk < 4; ++kk) {
      const int k0 = kk * 32;
      v16h b0 = load_b16(gwp + (32 * w +  0) * DIM + k0, DIM, lane);
      v16h b1 = load_b16(gwp + (32 * w + 16) * DIM + k0, DIM, lane);
#pragma unroll
      for (int mt = 0; mt < 4; ++mt) {
        v16h a = load_a32(sO + mt * 16 * DIM + k0, DIM, lane);
        acc[mt][0] = WMMA_F16(a, b0, acc[mt][0]);
        acc[mt][1] = WMMA_F16(a, b1, acc[mt][1]);
      }
    }
    float* op = out + base;
#pragma unroll
    for (int mt = 0; mt < 4; ++mt)
#pragma unroll
      for (int nt = 0; nt < 2; ++nt) {
        const int n  = 32 * w + nt * 16 + (lane & 15);
        const float pb = proj_b[n];
        const int m0 = mt * 16 + ((lane >> 4) << 3);
#pragma unroll
        for (int r = 0; r < 8; ++r) op[(m0 + r) * DIM + n] = acc[mt][nt][r] + pb;
      }
  }
}

// ---------------------------------------------------------------------------
// Launch
// ---------------------------------------------------------------------------

extern "C" void kernel_launch(void* const* d_in, const int* in_sizes, int n_in,
                              void* d_out, int out_size, void* d_ws, size_t ws_size,
                              hipStream_t stream) {
  (void)in_sizes; (void)n_in; (void)out_size; (void)ws_size;
  const float* x        = (const float*)d_in[0];
  const float* ctx      = (const float*)d_in[1];
  const float* mask     = (const float*)d_in[2];
  const float* q_w      = (const float*)d_in[3];
  const float* q_b      = (const float*)d_in[4];
  const float* kv_w     = (const float*)d_in[5];
  const float* v_b      = (const float*)d_in[6];
  const float* lscale   = (const float*)d_in[7];
  const float* cpb_w1   = (const float*)d_in[8];
  const float* cpb_b1   = (const float*)d_in[9];
  const float* cpb_w2   = (const float*)d_in[10];
  const float* proj_w   = (const float*)d_in[11];
  const float* proj_b   = (const float*)d_in[12];
  const float* rel_tab  = (const float*)d_in[13];
  const int*   rel_idx  = (const int*)d_in[14];

  char* ws = (char*)d_ws;
  _Float16* gwq    = (_Float16*)(ws + WS_GWQ);
  _Float16* gwk    = (_Float16*)(ws + WS_GWK);
  _Float16* gwv    = (_Float16*)(ws + WS_GWV);
  _Float16* gwp    = (_Float16*)(ws + WS_GWP);
  float*    scaleH = (float*)(ws + WS_SCALE);
  float*    tblw   = (float*)(ws + WS_TBL);
  float*    biasX  = (float*)(ws + WS_BIAS);

  prep_weights_kernel<<<64, 256, 0, stream>>>(q_w, kv_w, proj_w, gwq, gwk, gwv, gwp);
  prep_cpb_kernel<<<1, 256, 0, stream>>>(rel_tab, cpb_w1, cpb_b1, cpb_w2, lscale, tblw, scaleH);
  prep_bias_kernel<<<64, 256, 0, stream>>>(tblw, rel_idx, biasX);
  swin_attn_kernel<<<BWIN, 128, SMEM_BYTES, stream>>>(
      x, ctx, mask, q_b, v_b, gwq, gwk, gwv, gwp, proj_b, scaleH, biasX, (float*)d_out);
}
